// DGMGearnet_only_sequence_17033840295950
// MI455X (gfx1250) — matrix operand order repaired and checked by reading, not compiled, not run on hardware
//
#include <hip/hip_runtime.h>

// ---------------- problem constants ----------------
#define Nn     3072
#define Rr     7
#define ATTNR  2
#define Dd     64
#define Hh     4
#define DHd    16
#define WIN    128
#define BAND   257            // 2*WIN+1
#define TOPKk  16
#define RN     (Rr * Nn)      // 21504  (new_edge_list row length)
#define RD     (Rr * Dd)      // 448    (msg2 row length)

typedef __attribute__((ext_vector_type(16))) _Float16 v16h;
typedef __attribute__((ext_vector_type(8)))  _Float16 v8h;
typedef __attribute__((ext_vector_type(8)))  float    v8f;
typedef unsigned int u32x4 __attribute__((ext_vector_type(4)));
typedef int          i32x4 __attribute__((ext_vector_type(4)));
typedef int          i32x8 __attribute__((ext_vector_type(8)));

// ======================================================================
// TDM: 1-D contiguous global -> LDS copy via tensor descriptor (D#).
// Group0: count=1 | lds_addr | global_addr[56:0] | type=2 ("image").
// Group1: workgroup_mask=0, data_size=3 (8B), tensor_dim0=huge (no OOB),
//         tile_dim0 = n8 (8-byte units), tile_dim1/2 = 0 (1-D tile).
// Groups 2/3 zero (<=2-D tensor). Tracked by TENSORcnt.
// This toolchain (clang-23 / therock-10.0) uses the 6-arg builtin form.
// ======================================================================
__device__ inline void tdm_load_1d(unsigned lds_addr, const void* gptr, unsigned n8) {
  unsigned long long ga = (unsigned long long)gptr;
  u32x4 g0;
  g0[0] = 1u;                                                  // count=1, user D#
  g0[1] = lds_addr;                                            // LDS byte address
  g0[2] = (unsigned)(ga & 0xFFFFFFFFu);                        // global_addr[31:0]
  g0[3] = (unsigned)((ga >> 32) & 0x1FFFFFFu) | (2u << 30);    // [56:32] | type=2
  i32x8 g1;
  g1[0] = (int)(3u << 16);                                     // data_size=8B, mask=0
  g1[1] = (int)0xFFFF0000;                                     // tensor_dim0 lo16 (huge)
  g1[2] = (int)((1u << 16) | 0x7FFFu);                         // td0 hi16, tensor_dim1=1
  g1[3] = (int)((n8 & 0xFFFFu) << 16);                         // tile_dim0 = n8
  g1[4] = 0;                                                   // tile_dim1=0, tile_dim2=0
  g1[5] = 0x7FFFFFFF;                                          // tensor_dim0_stride lo
  g1[6] = 0;
  g1[7] = 0;
  i32x4 z4 = {0, 0, 0, 0};
  i32x8 z8 = {0, 0, 0, 0, 0, 0, 0, 0};
  __builtin_amdgcn_tensor_load_to_lds(g0, g1, z4, z4, z8, 0);
}

// ======================================================================
// f32 -> f16 elementwise convert
// ======================================================================
__global__ void f32_to_f16_kernel(const float* __restrict__ in, _Float16* __restrict__ out, int n) {
  int i = blockIdx.x * 256 + threadIdx.x;
  if (i < n) out[i] = (_Float16)in[i];
}

// ======================================================================
// Pack an f32 (K x 64) matrix into WMMA B-operand layout, f16:
// per (kb = K/32 tile, cb = col/16 tile): 32 lanes x 16 contiguous halfs.
// ======================================================================
__global__ void pack_b_kernel(const float* __restrict__ W, _Float16* __restrict__ P, int K) {
  int blk  = blockIdx.x;
  int kb   = blk >> 2, cb = blk & 3;
  int lane = threadIdx.x & 31;
  const float* p = W + (size_t)(kb * 32 + ((lane >> 4) << 3)) * Dd + cb * 16 + (lane & 15);
  _Float16* o = P + ((size_t)blk * 32 + lane) * 16;
#pragma unroll
  for (int t = 0; t < 16; ++t) o[t] = (_Float16)p[(size_t)(t + (t & 8)) * Dd];
}

// ======================================================================
// WMMA A fragment from f16 row-major: lanes 0-15 row M=lane, K{0..7,16..23};
// lanes 16-31 row M=lane-16, K{8..15,24..31}. Two 16B contiguous loads.
// ======================================================================
__device__ inline v16h load_a_h(const _Float16* A, int lda, int row0, int k0, int lane) {
  const _Float16* p = A + (size_t)(row0 + (lane & 15)) * lda + k0 + ((lane >> 4) << 3);
  v8h lo = *(const v8h*)p;
  v8h hi = *(const v8h*)(p + 16);
  return __builtin_shufflevector(lo, hi, 0, 1, 2, 3, 4, 5, 6, 7, 8, 9, 10, 11, 12, 13, 14, 15);
}

__device__ inline v16h load_b_lds(const _Float16* ldsB, int k0, int wave, int lane) {
  const _Float16* p = ldsB + (((size_t)(k0 >> 5) * 4 + wave) * 32 + lane) * 16;
  v8h lo = *(const v8h*)p;
  v8h hi = *(const v8h*)(p + 8);
  return __builtin_shufflevector(lo, hi, 0, 1, 2, 3, 4, 5, 6, 7, 8, 9, 10, 11, 12, 13, 14, 15);
}

// ======================================================================
// edge scatter: one block per edge (64 threads = feature row)
// ======================================================================
__global__ void edge_scatter_kernel(const float* __restrict__ x,  const float* __restrict__ ew,
                                    const int* __restrict__ ni,   const int* __restrict__ no,
                                    const int* __restrict__ rel,
                                    float* __restrict__ NEL, float* __restrict__ msg56,
                                    float* __restrict__ msg2) {
  int e = blockIdx.x;
  int d = threadIdx.x;
  int i = ni[e], j = no[e], r = rel[e];
  float w  = ew[e];
  float xv = w * x[(size_t)i * Dd + d];
  if (r >= 5) {
    atomicAdd(&msg56[((size_t)(r - 5) * Nn + j) * Dd + d], xv);
    if (d == 0) atomicAdd(&NEL[(size_t)i * RN + (size_t)(r - 5) * Nn + j], w);
  } else {
    atomicAdd(&msg2[(size_t)j * RD + (size_t)(r + 2) * Dd + d], xv);
    if (d == 0) atomicAdd(&NEL[(size_t)i * RN + (size_t)(r + 2) * Nn + j], w);
  }
}

// ======================================================================
// fused GEMM on f16 operands: C = A1@B1 (+ A2@B2) (+ bias), 64 output cols,
// 4 waves/block. Packed B weight regions are DMA'd once per block into LDS
// by the Tensor Data Mover (wave 0, s_wait_tensorcnt), then every wave reads
// its B fragments from LDS; A streams straight from global as b128 loads.
// Dynamic LDS = (k1+k2)*128 bytes.
// ======================================================================
__global__ void gemm2h_kernel(const _Float16* __restrict__ A1, int lda1,
                              const _Float16* __restrict__ B1, int k1,
                              const _Float16* __restrict__ A2, int lda2,
                              const _Float16* __restrict__ B2, int k2,
                              const float* __restrict__ bias,
                              float* __restrict__ C, int ldc) {
  extern __shared__ _Float16 ldsB[];
  int row0 = blockIdx.x * 16;
  int lane = threadIdx.x & 31;
  int wave = threadIdx.x >> 5;
  int col0 = wave * 16;

  if (threadIdx.x < 32) {   // one wave issues the TDM descriptors (EXEC ignored)
    tdm_load_1d((unsigned)(size_t)ldsB, B1, (unsigned)(k1 * 16));         // k1*128 B
    if (k2 > 0)
      tdm_load_1d((unsigned)(size_t)(ldsB + (size_t)k1 * 64), B2, (unsigned)(k2 * 16));
    __builtin_amdgcn_s_wait_tensorcnt(0);
  }
  __syncthreads();

  v8f acc = {};
  for (int k0 = 0; k0 < k1; k0 += 32) {
    __builtin_prefetch(A1 + (size_t)(row0 + (lane & 15)) * lda1 + k0 + 32, 0, 0);
    v16h a = load_a_h(A1, lda1, row0, k0, lane);
    v16h b = load_b_lds(ldsB, k0, wave, lane);
    acc = __builtin_amdgcn_wmma_f32_16x16x32_f16(false, a, false, b, (short)0, acc, false, false);
  }
  const _Float16* ldsB2 = ldsB + (size_t)k1 * 64;
  for (int k0 = 0; k0 < k2; k0 += 32) {
    v16h a = load_a_h(A2, lda2, row0, k0, lane);
    v16h b = load_b_lds(ldsB2, k0, wave, lane);
    acc = __builtin_amdgcn_wmma_f32_16x16x32_f16(false, a, false, b, (short)0, acc, false, false);
  }
  int n  = col0 + (lane & 15);
  int mo = (lane >> 4) << 3;                 // lanes>=16 hold rows M=8..15
  float bv = bias ? bias[n] : 0.0f;
#pragma unroll
  for (int v = 0; v < 8; ++v)
    C[(size_t)(row0 + mo + v) * ldc + n] = acc[v] + bv;
}

// ======================================================================
// BatchNorm over axis N + ReLU; optional f32 out, f16 out, column sum
// ======================================================================
__global__ void bn_relu_kernel(const float* __restrict__ in, float* __restrict__ outf,
                               _Float16* __restrict__ outh, float* __restrict__ gf) {
  int col = blockIdx.x;
  int tid = threadIdx.x;
  __shared__ float r1[256], r2[256];
  float s = 0.f, s2 = 0.f;
  for (int r = tid; r < Nn; r += 256) {
    float v = in[(size_t)r * Dd + col];
    s += v; s2 += v * v;
  }
  r1[tid] = s; r2[tid] = s2;
  __syncthreads();
  for (int o = 128; o > 0; o >>= 1) {
    if (tid < o) { r1[tid] += r1[tid + o]; r2[tid] += r2[tid + o]; }
    __syncthreads();
  }
  float mean = r1[0] * (1.0f / Nn);
  float var  = r2[0] * (1.0f / Nn) - mean * mean;
  float inv  = rsqrtf(var + 1e-5f);
  __syncthreads();
  float gs = 0.f;
  for (int r = tid; r < Nn; r += 256) {
    float v = (in[(size_t)r * Dd + col] - mean) * inv;
    v = fmaxf(v, 0.f);
    if (outf) outf[(size_t)r * Dd + col] = v;
    if (outh) outh[(size_t)r * Dd + col] = (_Float16)v;
    gs += v;
  }
  if (gf) {
    r1[tid] = gs;
    __syncthreads();
    for (int o = 128; o > 0; o >>= 1) {
      if (tid < o) r1[tid] += r1[tid + o];
      __syncthreads();
    }
    if (tid == 0) gf[col] = r1[0];
  }
}

// ======================================================================
// banded softmax + head-mean + exact top-16 threshold + max into NEL.
// Out-of-band logits are -2e9 -> expf underflows to 0 in f32, so the 257-wide
// band reproduces the reference softmax exactly. Band >= 129 >= 16 everywhere.
// block = (row n, attn relation r); 4 waves = 4 heads.
// ======================================================================
__global__ void attn_topk_kernel(const float* __restrict__ qb, const float* __restrict__ kb,
                                 float* __restrict__ NEL) {
  int n    = blockIdx.x;
  int r    = blockIdx.y;
  int wave = threadIdx.x >> 5;
  int lane = threadIdx.x & 31;
  __shared__ float ph[Hh][BAND];
  __shared__ float pm[BAND];
  __shared__ float thr_s;

  const float* qrow = qb + ((size_t)r * Nn + n) * Dd + wave * DHd;
  float q[DHd];
#pragma unroll
  for (int t = 0; t < DHd; ++t) q[t] = qrow[t];

  // scores/sqrt(DH)/TEMP = qk * 0.25 * 2 = qk * 0.5
  float sv[9];
  float smax = -3.0e38f;
#pragma unroll
  for (int it = 0; it < 9; ++it) {
    int t = lane + 32 * it;
    int m = n - WIN + t;
    float s = -3.0e38f;
    if (t < BAND && m >= 0 && m < Nn) {
      const float* krow = kb + ((size_t)r * Nn + m) * Dd + wave * DHd;
      float acc = 0.f;
#pragma unroll
      for (int u = 0; u < DHd; ++u) acc += q[u] * krow[u];
      s = 0.5f * acc;
    }
    sv[it] = s;
    smax = fmaxf(smax, s);
  }
#pragma unroll
  for (int o = 16; o > 0; o >>= 1) smax = fmaxf(smax, __shfl_xor(smax, o, 32));
  float esum = 0.f;
#pragma unroll
  for (int it = 0; it < 9; ++it) {
    float e = (sv[it] > -1.0e37f) ? __expf(sv[it] - smax) : 0.f;
    sv[it] = e;
    esum += e;
  }
#pragma unroll
  for (int o = 16; o > 0; o >>= 1) esum += __shfl_xor(esum, o, 32);
  float pinv = 1.0f / esum;
#pragma unroll
  for (int it = 0; it < 9; ++it) {
    int t = lane + 32 * it;
    if (t < BAND) ph[wave][t] = sv[it] * pinv;
  }
  __syncthreads();
  for (int t = threadIdx.x; t < BAND; t += 128)
    pm[t] = 0.25f * (ph[0][t] + ph[1][t] + ph[2][t] + ph[3][t]);
  __syncthreads();

  // exact 16-th largest (duplicates counted): 16 argmax-remove passes on wave 0
  if (wave == 0) {
    float sel[9];
#pragma unroll
    for (int it = 0; it < 9; ++it) {
      int t = lane + 32 * it;
      sel[it] = (t < BAND) ? pm[t] : -1.0e30f;
    }
    float th = 0.f;
    for (int pass = 0; pass < TOPKk; ++pass) {
      float bv = sel[0]; int bt = lane;
#pragma unroll
      for (int it = 1; it < 9; ++it) {
        int t = lane + 32 * it;
        if (sel[it] > bv) { bv = sel[it]; bt = t; }
      }
#pragma unroll
      for (int o = 16; o > 0; o >>= 1) {
        float ov = __shfl_xor(bv, o, 32);
        int   ot = __shfl_xor(bt, o, 32);
        if (ov > bv || (ov == bv && ot < bt)) { bv = ov; bt = ot; }
      }
      if ((bt & 31) == lane) sel[bt >> 5] = -1.0e30f;  // remove exactly one element
      th = bv;
    }
    if (lane == 0) thr_s = th;
  }
  __syncthreads();
  float th = thr_s;
  // attn_output = max(adjacency, S): adjacency already lives in NEL; S==0 outside
  // the band, so only band cells with p>=th need the max. Single writer per cell.
  for (int t = threadIdx.x; t < BAND; t += 128) {
    int m = n - WIN + t;
    if (m >= 0 && m < Nn) {
      float p = pm[t];
      if (p >= th) {
        size_t idx = (size_t)n * RN + (size_t)r * Nn + m;
        float old = NEL[idx];
        if (p > old) NEL[idx] = p;
      }
    }
  }
}

// ======================================================================
// msg2 dense relations 0,1: msg2[:, c*64:+64] = NEL[:, c*N:+N]^T @ x.
// Transposed NEL tile (32 K-rows x 16 j-cols, f32) staged into LDS with async
// global->LDS copies (wave 0 issues 4x B128), s_wait_asynccnt + barrier.
// B operand is x pre-packed in WMMA layout (one 32B load per step).
// ======================================================================
__global__ void msg2_dense_kernel(const float* __restrict__ NEL, const v16h* __restrict__ xpack,
                                  float* __restrict__ msg2) {
  int j0   = blockIdx.x * 16;
  int c    = blockIdx.y;
  int lane = threadIdx.x & 31;
  int wave = threadIdx.x >> 5;
  int col0 = wave * 16;
  __shared__ float tile[32][16];     // [K row][j col]
  v8f acc = {};
  for (int i0 = 0; i0 < Nn; i0 += 32) {
    if (wave == 0) {
      unsigned ldsbase = (unsigned)(size_t)(&tile[0][0]);
#pragma unroll
      for (int s = 0; s < 4; ++s) {
        int li  = s * 32 + lane;
        int row = li >> 2;            // 0..31 K row
        int seg = (li & 3) << 2;      // 0,4,8,12 j offset
        unsigned long long g = (unsigned long long)(const void*)
            &NEL[(size_t)(i0 + row) * RN + (size_t)c * Nn + j0 + seg];
        unsigned lo = ldsbase + (unsigned)((row * 16 + seg) * 4);
        asm volatile("global_load_async_to_lds_b128 %0, %1, off"
                     :: "v"(lo), "v"(g) : "memory");
      }
      asm volatile("s_wait_asynccnt 0x0" ::: "memory");
    }
    __syncthreads();
    v16h a;
    {
      int jr  = lane & 15;
      int kbv = (lane >> 4) << 3;
#pragma unroll
      for (int t = 0; t < 8; ++t) {
        a[t]     = (_Float16)tile[kbv + t][jr];
        a[t + 8] = (_Float16)tile[kbv + 16 + t][jr];
      }
    }
    v16h b = xpack[((size_t)(i0 >> 5) * 4 + wave) * 32 + lane];
    acc = __builtin_amdgcn_wmma_f32_16x16x32_f16(false, a, false, b, (short)0, acc, false, false);
    __syncthreads();
  }
  int d  = col0 + (lane & 15);
  int mo = (lane >> 4) << 3;
#pragma unroll
  for (int v = 0; v < 8; ++v)
    msg2[(size_t)(j0 + mo + v) * RD + (size_t)c * Dd + d] = acc[v];
}

// ======================================================================
// host-side orchestration
// ======================================================================
extern "C" void kernel_launch(void* const* d_in, const int* in_sizes, int n_in,
                              void* d_out, int out_size, void* d_ws, size_t ws_size,
                              hipStream_t stream) {
  (void)in_sizes; (void)n_in; (void)out_size; (void)ws_size;
  const float* x        = (const float*)d_in[0];
  const float* ew       = (const float*)d_in[1];
  const float* W_rel_s  = (const float*)d_in[2];
  const float* W_self_s = (const float*)d_in[3];
  const float* b_s      = (const float*)d_in[4];
  const float* Wq       = (const float*)d_in[5];
  const float* Wk       = (const float*)d_in[6];
  const float* W_msg    = (const float*)d_in[7];
  const float* W_self_g = (const float*)d_in[8];
  const float* b_g      = (const float*)d_in[9];
  const int*   ni       = (const int*)d_in[10];
  const int*   no       = (const int*)d_in[11];
  const int*   rel      = (const int*)d_in[12];

  float* out    = (float*)d_out;
  float* gf     = out;                          // (1,64)
  float* hidden = out + 64;                     // (3072,64)
  float* NEL    = out + 64 + (size_t)Nn * Dd;   // (3072,21504)

  // ---- f32 workspace ----
  float* ws    = (float*)d_ws;
  float* msg56 = ws;                            // 2*3072*64   = 393216
  float* pre56 = msg56 + 2 * Nn * Dd;           // 393216
  float* qb    = pre56 + 2 * Nn * Dd;           // 393216
  float* kbuf  = qb    + 2 * Nn * Dd;           // 393216
  float* msg2  = kbuf  + 2 * Nn * Dd;           // 3072*448 = 1376256
  float* tmp_h = msg2  + (size_t)Nn * RD;       // 196608
  // ---- f16 workspace (starts 16B-aligned) ----
  _Float16* hb      = (_Float16*)(tmp_h + (size_t)Nn * Dd);
  _Float16* x_h     = hb;                                 // 196608 halfs
  _Float16* msg56_h = x_h     + (size_t)Nn * Dd;          // 393216
  _Float16* attn_h  = msg56_h + 2 * (size_t)Nn * Dd;      // 393216
  _Float16* msg2_h  = attn_h  + 2 * (size_t)Nn * Dd;      // 1376256
  _Float16* xpack   = msg2_h  + (size_t)Nn * RD;          // 196608
  _Float16* pWrel5  = xpack   + (size_t)Nn * Dd;          // 4096 each below
  _Float16* pWrel6  = pWrel5  + 4096;
  _Float16* pWselfS = pWrel6  + 4096;
  _Float16* pWq0    = pWselfS + 4096;
  _Float16* pWq1    = pWq0    + 4096;
  _Float16* pWk0    = pWq1    + 4096;
  _Float16* pWk1    = pWk0    + 4096;
  _Float16* pWselfG = pWk1    + 4096;
  _Float16* pWmsg   = pWselfG + 4096;                     // 28672

  // zero accumulation targets (memset nodes are graph-capturable)
  (void)hipMemsetAsync(NEL,   0, (size_t)Nn * RN * sizeof(float), stream);
  (void)hipMemsetAsync(msg56, 0, (size_t)2 * Nn * Dd * sizeof(float), stream);
  (void)hipMemsetAsync(msg2,  0, (size_t)Nn * RD * sizeof(float), stream);

  // 0) one-time operand conversion / packing (f16, WMMA layouts)
  f32_to_f16_kernel<<<(Nn * Dd + 255) / 256, 256, 0, stream>>>(x, x_h, Nn * Dd);
  pack_b_kernel<<<8,   32, 0, stream>>>(W_rel_s + 5 * Dd * Dd, pWrel5, Dd);
  pack_b_kernel<<<8,   32, 0, stream>>>(W_rel_s + 6 * Dd * Dd, pWrel6, Dd);
  pack_b_kernel<<<8,   32, 0, stream>>>(W_self_s, pWselfS, Dd);
  pack_b_kernel<<<8,   32, 0, stream>>>(Wq,               pWq0, Dd);
  pack_b_kernel<<<8,   32, 0, stream>>>(Wq + Dd * Dd,     pWq1, Dd);
  pack_b_kernel<<<8,   32, 0, stream>>>(Wk,               pWk0, Dd);
  pack_b_kernel<<<8,   32, 0, stream>>>(Wk + Dd * Dd,     pWk1, Dd);
  pack_b_kernel<<<8,   32, 0, stream>>>(W_self_g, pWselfG, Dd);
  pack_b_kernel<<<56,  32, 0, stream>>>(W_msg, pWmsg, RD);
  pack_b_kernel<<<384, 32, 0, stream>>>(x, xpack, Nn);   // x as B operand (K=3072)

  // 1) sparse scatter: adjacency into d_out, msg for relations 5,6, sparse msg2 blocks
  edge_scatter_kernel<<<Nn * 16, Dd, 0, stream>>>(x, ew, ni, no, rel, NEL, msg56, msg2);
  f32_to_f16_kernel<<<(2 * Nn * Dd + 255) / 256, 256, 0, stream>>>(msg56, msg56_h, 2 * Nn * Dd);

  // 2) rel_out (only relations 5,6 are consumed): pre = msg@W_rel + x@W_self + b_s
  const _Float16* pWrel[2] = {pWrel5, pWrel6};
  for (int r = 0; r < ATTNR; ++r)
    gemm2h_kernel<<<Nn / 16, 128, (Dd + Dd) * 128, stream>>>(
        msg56_h + (size_t)r * Nn * Dd, Dd, pWrel[r], Dd,
        x_h, Dd, pWselfS, Dd, b_s, pre56 + (size_t)r * Nn * Dd, Dd);
  for (int r = 0; r < ATTNR; ++r)
    bn_relu_kernel<<<Dd, 256, 0, stream>>>(pre56 + (size_t)r * Nn * Dd, nullptr,
                                           attn_h + (size_t)r * Nn * Dd, nullptr);

  // 3) q, k projections
  const _Float16* pWqv[2] = {pWq0, pWq1};
  const _Float16* pWkv[2] = {pWk0, pWk1};
  for (int r = 0; r < ATTNR; ++r) {
    gemm2h_kernel<<<Nn / 16, 128, Dd * 128, stream>>>(
        attn_h + (size_t)r * Nn * Dd, Dd, pWqv[r], Dd,
        nullptr, 0, nullptr, 0, nullptr, qb + (size_t)r * Nn * Dd, Dd);
    gemm2h_kernel<<<Nn / 16, 128, Dd * 128, stream>>>(
        attn_h + (size_t)r * Nn * Dd, Dd, pWkv[r], Dd,
        nullptr, 0, nullptr, 0, nullptr, kbuf + (size_t)r * Nn * Dd, Dd);
  }

  // 4) banded softmax + top-16 + max into new_edge_list
  attn_topk_kernel<<<dim3(Nn, ATTNR), 128, 0, stream>>>(qb, kbuf, NEL);

  // 5) dense second aggregation for relations 0,1 (reads the maxed NEL, async->LDS)
  msg2_dense_kernel<<<dim3(Nn / 16, 2), 128, 0, stream>>>(NEL, (const v16h*)xpack, msg2);
  f32_to_f16_kernel<<<((int)(Nn * RD) + 255) / 256, 256, 0, stream>>>(msg2, msg2_h, Nn * RD);

  // 6) hidden pre-activation: msg2@W_msg + x@W_self_g + b_g  (TDM stages 64KB of B)
  gemm2h_kernel<<<Nn / 16, 128, (RD + Dd) * 128, stream>>>(msg2_h, RD, pWmsg, RD,
                                                           x_h, Dd, pWselfG, Dd, b_g, tmp_h, Dd);

  // 7) BN + ReLU + graph_feature column sum
  bn_relu_kernel<<<Dd, 256, 0, stream>>>(tmp_h, hidden, nullptr, gf);
}
